// ThreeLayerResidualGAT_429496730272
// MI455X (gfx1250) — matrix operand order repaired and checked by reading, not compiled, not run on hardware
//
#include <hip/hip_runtime.h>
#include <hip/hip_bf16.h>
#include <cstdint>

// ---------------- problem constants ----------------
#define NN      50000      // nodes
#define F0_     128
#define EE      400000     // edges (before self loops)
#define DD      1024       // H1*C1
#define H1_     4
#define C1_     256
#define H3_     6
#define NC_     40
#define D3_     (H3_*NC_)  // 240
#define NEG_SLOPE 0.2f

typedef __bf16 bf16;
typedef __attribute__((ext_vector_type(16))) __bf16 v16bf;
typedef __attribute__((ext_vector_type(8)))  __bf16 v8bf;
typedef __attribute__((ext_vector_type(8)))  float  v8f;
typedef int v4i __attribute__((vector_size(16)));

// CDNA5 async global->LDS path (guarded: falls back to VGPR staging if absent)
#if __has_builtin(__builtin_amdgcn_global_load_async_to_lds_b128)
#define USE_ASYNC_LDS 1
#else
#define USE_ASYNC_LDS 0
#endif

#if USE_ASYNC_LDS
// one 16-byte chunk, global -> LDS, tracked by ASYNCcnt
__device__ inline void async_copy16(const bf16* g, bf16* l) {
  __builtin_amdgcn_global_load_async_to_lds_b128(
      (__attribute__((address_space(1))) v4i*)(void*)const_cast<bf16*>(g),
      (__attribute__((address_space(3))) v4i*)(void*)l,
      0, 0);
}
#endif

__device__ inline void wait_async_zero() {
#if __has_builtin(__builtin_amdgcn_s_wait_asynccnt)
  __builtin_amdgcn_s_wait_asynccnt(0);
#else
  asm volatile("s_wait_asynccnt 0x0" ::: "memory");
#endif
}

// ---------------- workspace layout (bytes, 256B aligned) ----------------
// total ~475 MB
constexpr size_t OFF_HB   = 0;                          // bf16 activation in  [N,1024]
constexpr size_t OFF_H    = OFF_HB  + 102400000;        // bf16 GAT linear out [N,1024]
constexpr size_t OFF_RES  = OFF_H   + 102400000;        // f32 residual/acc    [N,1024]
constexpr size_t OFF_WB   = OFF_RES + 204800000;        // bf16 weight staging [1024,1024]
constexpr size_t OFF_AS   = OFF_WB  + 2097152;          // f32 [N,6]
constexpr size_t OFF_AD   = OFF_AS  + 1200128;
constexpr size_t OFF_MX   = OFF_AD  + 1200128;
constexpr size_t OFF_DEN  = OFF_MX  + 1200128;
constexpr size_t OFF_P    = OFF_DEN + 1200128;          // f32 [E+N,6]
constexpr size_t OFF_ACC3 = OFF_P   + 10800128;         // f32 [N,240]

// ---------------- small utility kernels ----------------
__global__ void f32_to_bf16_kernel(const float* __restrict__ in, bf16* __restrict__ out, long n) {
  long i = (long)blockIdx.x * blockDim.x + threadIdx.x;
  if (i < n) out[i] = (bf16)in[i];
}

__global__ void fill_f32_kernel(float* __restrict__ p, long n, float v) {
  long i = (long)blockIdx.x * blockDim.x + threadIdx.x;
  if (i < n) p[i] = v;
}

__global__ void add_bias2_kernel(float* __restrict__ res, const float* __restrict__ b,
                                 const float* __restrict__ lb, long n, int Dl) {
  long i = (long)blockIdx.x * blockDim.x + threadIdx.x;
  if (i < n) { int j = (int)(i % Dl); res[i] += b[j] + lb[j]; }
}

__global__ void finalize_elu_bf16_kernel(const float* __restrict__ res, bf16* __restrict__ xout, long n) {
  long i = (long)blockIdx.x * blockDim.x + threadIdx.x;
  if (i < n) {
    float v = res[i];
    float e = v > 0.f ? v : (__expf(v) - 1.0f);
    xout[i] = (bf16)e;
  }
}

// out[n,40] = mean over 6 heads of acc3[n,240] + res3[n,40] (res3 already biased)
__global__ void finalize3_kernel(const float* __restrict__ acc3, const float* __restrict__ res3,
                                 float* __restrict__ out, int n) {
  int i = blockIdx.x * blockDim.x + threadIdx.x;
  if (i >= n * NC_) return;
  int node = i / NC_, c = i % NC_;
  float s = 0.f;
#pragma unroll
  for (int h = 0; h < H3_; h++) s += acc3[(size_t)node * D3_ + h * NC_ + c];
  out[i] = s * (1.0f / H3_) + res3[i];
}

// ---------------- WMMA GEMM: C[M,Nn] = A[M,K](bf16) x B[K,Nn](bf16) ----------------
// block = 256 threads (8 waves), block tile 128x64, wave tile 32x32 (4 wmma / K-step)
// NOTE: requires Nn % 8 == 0 and K % 32 == 0 (true for 1024/240/40 and 128/1024).
template <bool OUT_BF16>
__global__ __launch_bounds__(256)
void gemm_bf16_wmma(const bf16* __restrict__ A, const bf16* __restrict__ B,
                    void* __restrict__ Cout, int M, int Nn, int K) {
  __shared__ __align__(16) bf16 As[128 * 40];   // 128 rows x 32 K (stride 40 keeps 16B align)
  __shared__ __align__(16) bf16 Bt[64 * 40];    // transposed: 64 N-cols x 32 K

  const int tid  = threadIdx.x;
  const int lane = tid & 31;
  const int wid  = tid >> 5;
  const int rowBase = blockIdx.y * 128;
  const int colBase = blockIdx.x * 64;

  const int wy = wid >> 1, wx = wid & 1;
  const int tm0 = wy * 32, tm1 = wy * 32 + 16;
  const int tn0 = wx * 32, tn1 = wx * 32 + 16;
  const int half = lane >> 4;          // 0/1: which 16-lane half
  const int ml = lane & 15;            // A row within tile / C column
  const int kb = half * 16;            // unused; B fragments read K at half*8 and 16 + half*8

  // A-stage mapping: each thread owns one 16-element (32B) row segment
  const int ar   = tid >> 1;           // 0..127
  const int aseg = (tid & 1) * 16;     // 0 or 16
  const int agrow = rowBase + ar;
  const bf16* aSrcRow = A + (size_t)agrow * K + aseg;
  bf16* aDstLds = &As[ar * 40 + aseg];

  v8f c00 = {}, c01 = {}, c10 = {}, c11 = {};

  for (int kt = 0; kt < K; kt += 32) {
    // --- stage A tile: 128x32 ---
    if (agrow < M) {
      const bf16* src = aSrcRow + kt;
#if USE_ASYNC_LDS
      async_copy16(src,     aDstLds);
      async_copy16(src + 8, aDstLds + 8);
#else
      v8bf v0 = *reinterpret_cast<const v8bf*>(src);
      v8bf v1 = *reinterpret_cast<const v8bf*>(src + 8);
      *reinterpret_cast<v8bf*>(aDstLds)     = v0;
      *reinterpret_cast<v8bf*>(aDstLds + 8) = v1;
#endif
      if (kt + 32 < K) __builtin_prefetch(src + 32, 0, 1);   // global_prefetch next K slab
    }
    // --- stage B tile (K x 64) transposed into Bt[n][k]; Nn%8==0 so segments are all-or-nothing ---
    {
      int k = tid >> 3;                // 0..31
      int nseg = (tid & 7) * 8;        // 0..56
      v8bf v = {};
      if (colBase + nseg < Nn)
        v = *reinterpret_cast<const v8bf*>(B + (size_t)(kt + k) * Nn + colBase + nseg);
#pragma unroll
      for (int j = 0; j < 8; j++) Bt[(nseg + j) * 40 + k] = v[j];
    }
#if USE_ASYNC_LDS
    wait_async_zero();
#endif
    __syncthreads();

    // --- fragments (ISA 16-bit A 16x32 / B 32x16 layouts) ---
    v8bf a0lo = *reinterpret_cast<const v8bf*>(&As[(tm0 + ml) * 40 + half * 8]);
    v8bf a0hi = *reinterpret_cast<const v8bf*>(&As[(tm0 + ml) * 40 + 16 + half * 8]);
    v8bf a1lo = *reinterpret_cast<const v8bf*>(&As[(tm1 + ml) * 40 + half * 8]);
    v8bf a1hi = *reinterpret_cast<const v8bf*>(&As[(tm1 + ml) * 40 + 16 + half * 8]);
    v8bf b0lo = *reinterpret_cast<const v8bf*>(&Bt[(tn0 + ml) * 40 + half * 8]);
    v8bf b0hi = *reinterpret_cast<const v8bf*>(&Bt[(tn0 + ml) * 40 + 16 + half * 8]);
    v8bf b1lo = *reinterpret_cast<const v8bf*>(&Bt[(tn1 + ml) * 40 + half * 8]);
    v8bf b1hi = *reinterpret_cast<const v8bf*>(&Bt[(tn1 + ml) * 40 + 16 + half * 8]);

    v16bf a0 = __builtin_shufflevector(a0lo, a0hi, 0,1,2,3,4,5,6,7,8,9,10,11,12,13,14,15);
    v16bf a1 = __builtin_shufflevector(a1lo, a1hi, 0,1,2,3,4,5,6,7,8,9,10,11,12,13,14,15);
    v16bf b0 = __builtin_shufflevector(b0lo, b0hi, 0,1,2,3,4,5,6,7,8,9,10,11,12,13,14,15);
    v16bf b1 = __builtin_shufflevector(b1lo, b1hi, 0,1,2,3,4,5,6,7,8,9,10,11,12,13,14,15);

    c00 = __builtin_amdgcn_wmma_f32_16x16x32_bf16(false, a0, false, b0, (short)0, c00, false, false);
    c01 = __builtin_amdgcn_wmma_f32_16x16x32_bf16(false, a0, false, b1, (short)0, c01, false, false);
    c10 = __builtin_amdgcn_wmma_f32_16x16x32_bf16(false, a1, false, b0, (short)0, c10, false, false);
    c11 = __builtin_amdgcn_wmma_f32_16x16x32_bf16(false, a1, false, b1, (short)0, c11, false, false);
    __syncthreads();
  }

  // --- epilogue: C layout lane n = lane&15, VGPR r -> M = (half*8)+r ---
  auto store_tile = [&](const v8f& c, int tm, int tn) {
    int col = colBase + tn + ml;
    int mstart = rowBase + tm + half * 8;
#pragma unroll
    for (int r = 0; r < 8; r++) {
      int m = mstart + r;
      if (m < M && col < Nn) {
        if (OUT_BF16) ((bf16*)Cout)[(size_t)m * Nn + col] = (bf16)c[r];
        else          ((float*)Cout)[(size_t)m * Nn + col] = c[r];
      }
    }
  };
  store_tile(c00, tm0, tn0);
  store_tile(c01, tm0, tn1);
  store_tile(c10, tm1, tn0);
  store_tile(c11, tm1, tn1);
}

// ---------------- attention: a_s/a_d = einsum('nhc,hc->nh') ----------------
// one wave32 per (node, head)
__global__ __launch_bounds__(256)
void att_scores_kernel(const bf16* __restrict__ h, const float* __restrict__ a_src,
                       const float* __restrict__ a_dst, float* __restrict__ as,
                       float* __restrict__ ad, int n, int H, int C) {
  int wg = blockIdx.x * (blockDim.x >> 5) + (threadIdx.x >> 5);
  int lane = threadIdx.x & 31;
  if (wg >= n * H) return;  // uniform across the wave
  int node = wg / H, hh = wg % H;
  const bf16*  row = h + (size_t)node * H * C + hh * C;
  const float* asr = a_src + hh * C;
  const float* adr = a_dst + hh * C;
  float s = 0.f, d = 0.f;
  for (int c = lane; c < C; c += 32) {
    float v = (float)row[c];
    s += v * asr[c];
    d += v * adr[c];
  }
#pragma unroll
  for (int off = 16; off > 0; off >>= 1) {
    s += __shfl_xor(s, off, 32);
    d += __shfl_xor(d, off, 32);
  }
  if (lane == 0) { as[wg] = s; ad[wg] = d; }
}

// monotone float max via int-bits atomics
__device__ inline void atomicMaxF(float* addr, float v) {
  if (v >= 0.0f) atomicMax((int*)addr, __float_as_int(v));
  else           atomicMin((unsigned int*)addr, (unsigned int)__float_as_int(v));
}

__global__ void edge_max_kernel(const int* __restrict__ ei, const float* __restrict__ as,
                                const float* __restrict__ ad, float* __restrict__ mx,
                                int E, int n, int H) {
  int i = blockIdx.x * blockDim.x + threadIdx.x;
  if (i >= E + n) return;
  int src = (i < E) ? ei[i]     : (i - E);
  int dst = (i < E) ? ei[E + i] : (i - E);
  for (int h = 0; h < H; h++) {
    float e = as[src * H + h] + ad[dst * H + h];
    e = e > 0.f ? e : NEG_SLOPE * e;
    atomicMaxF(&mx[dst * H + h], e);
  }
}

__global__ void edge_exp_kernel(const int* __restrict__ ei, const float* __restrict__ as,
                                const float* __restrict__ ad, const float* __restrict__ mx,
                                float* __restrict__ den, float* __restrict__ p,
                                int E, int n, int H) {
  int i = blockIdx.x * blockDim.x + threadIdx.x;
  if (i >= E + n) return;
  int src = (i < E) ? ei[i]     : (i - E);
  int dst = (i < E) ? ei[E + i] : (i - E);
  for (int h = 0; h < H; h++) {
    float e = as[src * H + h] + ad[dst * H + h];
    e = e > 0.f ? e : NEG_SLOPE * e;
    float pe = __expf(e - mx[dst * H + h]);
    p[(size_t)i * H + h] = pe;
    atomicAdd(&den[dst * H + h], pe);
  }
}

__global__ void normalize_alpha_kernel(const int* __restrict__ ei, float* __restrict__ p,
                                       const float* __restrict__ den, float* __restrict__ outAlpha,
                                       int E, int n, int H) {
  int i = blockIdx.x * blockDim.x + threadIdx.x;
  if (i >= E + n) return;
  int dst = (i < E) ? ei[E + i] : (i - E);
  for (int h = 0; h < H; h++) {
    float a = p[(size_t)i * H + h] / (den[dst * H + h] + 1e-16f);
    p[(size_t)i * H + h] = a;
    if (outAlpha) outAlpha[(size_t)i * H + h] = a;
  }
}

// one block per edge: acc[dst] += alpha_h * h[src]
__global__ __launch_bounds__(256)
void scatter_msg_kernel(const int* __restrict__ ei, const bf16* __restrict__ h,
                        const float* __restrict__ alpha, float* __restrict__ acc,
                        int E, int n, int H, int C) {
  int i = blockIdx.x;
  int Dl = H * C;
  __shared__ float sal[8];
  __shared__ int s_src, s_dst;
  if (threadIdx.x == 0) {
    s_src = (i < E) ? ei[i]     : (i - E);
    s_dst = (i < E) ? ei[E + i] : (i - E);
  }
  if (threadIdx.x < H) sal[threadIdx.x] = alpha[(size_t)i * H + threadIdx.x];
  __syncthreads();
  int src = s_src, dst = s_dst;
  for (int c = threadIdx.x; c < Dl; c += blockDim.x) {
    int hh = c / C;
    float v = (float)h[(size_t)src * Dl + c] * sal[hh];
    atomicAdd(&acc[(size_t)dst * Dl + c], v);
  }
}

// ---------------- driver ----------------
static inline int cdiv(long a, long b) { return (int)((a + b - 1) / b); }

extern "C" void kernel_launch(void* const* d_in, const int* in_sizes, int n_in,
                              void* d_out, int out_size, void* d_ws, size_t ws_size,
                              hipStream_t stream) {
  const float* x   = (const float*)d_in[0];
  const int*   ei  = (const int*)  d_in[1];
  const float* W1  = (const float*)d_in[2];
  const float* a1s = (const float*)d_in[3];
  const float* a1d = (const float*)d_in[4];
  const float* b1  = (const float*)d_in[5];
  const float* lw1 = (const float*)d_in[6];
  const float* lb1 = (const float*)d_in[7];
  const float* W2  = (const float*)d_in[8];
  const float* a2s = (const float*)d_in[9];
  const float* a2d = (const float*)d_in[10];
  const float* b2  = (const float*)d_in[11];
  const float* lw2 = (const float*)d_in[12];
  const float* lb2 = (const float*)d_in[13];
  const float* W3  = (const float*)d_in[14];
  const float* a3s = (const float*)d_in[15];
  const float* a3d = (const float*)d_in[16];
  const float* b3  = (const float*)d_in[17];
  const float* lw3 = (const float*)d_in[18];
  const float* lb3 = (const float*)d_in[19];

  float* out       = (float*)d_out;               // [N,40]
  float* out_alpha = out + (size_t)NN * NC_;      // [E+N,4]

  char* ws = (char*)d_ws;
  bf16*  HB   = (bf16*)(ws + OFF_HB);
  bf16*  Hb   = (bf16*)(ws + OFF_H);
  float* RES  = (float*)(ws + OFF_RES);
  bf16*  WB   = (bf16*)(ws + OFF_WB);
  float* AS   = (float*)(ws + OFF_AS);
  float* AD   = (float*)(ws + OFF_AD);
  float* MX   = (float*)(ws + OFF_MX);
  float* DEN  = (float*)(ws + OFF_DEN);
  float* P    = (float*)(ws + OFF_P);
  float* ACC3 = (float*)(ws + OFF_ACC3);

  const int EN = EE + NN;
  const dim3 blk(256);

  // ================= layer 1 (F0 -> 4x256 concat) =================
  f32_to_bf16_kernel<<<cdiv((long)NN * F0_, 256), blk, 0, stream>>>(x, HB, (long)NN * F0_);
  f32_to_bf16_kernel<<<cdiv((long)F0_ * DD, 256), blk, 0, stream>>>(W1, WB, (long)F0_ * DD);
  {
    dim3 g(cdiv(DD, 64), cdiv(NN, 128));
    gemm_bf16_wmma<true><<<g, blk, 0, stream>>>(HB, WB, Hb, NN, DD, F0_);
  }
  f32_to_bf16_kernel<<<cdiv((long)F0_ * DD, 256), blk, 0, stream>>>(lw1, WB, (long)F0_ * DD);
  {
    dim3 g(cdiv(DD, 64), cdiv(NN, 128));
    gemm_bf16_wmma<false><<<g, blk, 0, stream>>>(HB, WB, RES, NN, DD, F0_);
  }
  att_scores_kernel<<<cdiv((long)NN * H1_, 8), blk, 0, stream>>>(Hb, a1s, a1d, AS, AD, NN, H1_, C1_);
  fill_f32_kernel<<<cdiv((long)NN * H1_, 256), blk, 0, stream>>>(MX, (long)NN * H1_, -INFINITY);
  fill_f32_kernel<<<cdiv((long)NN * H1_, 256), blk, 0, stream>>>(DEN, (long)NN * H1_, 0.f);
  edge_max_kernel<<<cdiv(EN, 256), blk, 0, stream>>>(ei, AS, AD, MX, EE, NN, H1_);
  edge_exp_kernel<<<cdiv(EN, 256), blk, 0, stream>>>(ei, AS, AD, MX, DEN, P, EE, NN, H1_);
  normalize_alpha_kernel<<<cdiv(EN, 256), blk, 0, stream>>>(ei, P, DEN, out_alpha, EE, NN, H1_);
  add_bias2_kernel<<<cdiv((long)NN * DD, 256), blk, 0, stream>>>(RES, b1, lb1, (long)NN * DD, DD);
  scatter_msg_kernel<<<EN, blk, 0, stream>>>(ei, Hb, P, RES, EE, NN, H1_, C1_);
  finalize_elu_bf16_kernel<<<cdiv((long)NN * DD, 256), blk, 0, stream>>>(RES, HB, (long)NN * DD);  // x1 -> HB

  // ================= layer 2 (1024 -> 4x256 concat) =================
  f32_to_bf16_kernel<<<cdiv((long)DD * DD, 256), blk, 0, stream>>>(W2, WB, (long)DD * DD);
  {
    dim3 g(cdiv(DD, 64), cdiv(NN, 128));
    gemm_bf16_wmma<true><<<g, blk, 0, stream>>>(HB, WB, Hb, NN, DD, DD);
  }
  f32_to_bf16_kernel<<<cdiv((long)DD * DD, 256), blk, 0, stream>>>(lw2, WB, (long)DD * DD);
  {
    dim3 g(cdiv(DD, 64), cdiv(NN, 128));
    gemm_bf16_wmma<false><<<g, blk, 0, stream>>>(HB, WB, RES, NN, DD, DD);
  }
  att_scores_kernel<<<cdiv((long)NN * H1_, 8), blk, 0, stream>>>(Hb, a2s, a2d, AS, AD, NN, H1_, C1_);
  fill_f32_kernel<<<cdiv((long)NN * H1_, 256), blk, 0, stream>>>(MX, (long)NN * H1_, -INFINITY);
  fill_f32_kernel<<<cdiv((long)NN * H1_, 256), blk, 0, stream>>>(DEN, (long)NN * H1_, 0.f);
  edge_max_kernel<<<cdiv(EN, 256), blk, 0, stream>>>(ei, AS, AD, MX, EE, NN, H1_);
  edge_exp_kernel<<<cdiv(EN, 256), blk, 0, stream>>>(ei, AS, AD, MX, DEN, P, EE, NN, H1_);
  normalize_alpha_kernel<<<cdiv(EN, 256), blk, 0, stream>>>(ei, P, DEN, nullptr, EE, NN, H1_);
  add_bias2_kernel<<<cdiv((long)NN * DD, 256), blk, 0, stream>>>(RES, b2, lb2, (long)NN * DD, DD);
  scatter_msg_kernel<<<EN, blk, 0, stream>>>(ei, Hb, P, RES, EE, NN, H1_, C1_);
  finalize_elu_bf16_kernel<<<cdiv((long)NN * DD, 256), blk, 0, stream>>>(RES, HB, (long)NN * DD);  // x2 -> HB

  // ================= layer 3 (1024 -> 6 heads x 40, mean) =================
  f32_to_bf16_kernel<<<cdiv((long)DD * D3_, 256), blk, 0, stream>>>(W3, WB, (long)DD * D3_);
  {
    dim3 g(cdiv(D3_, 64), cdiv(NN, 128));
    gemm_bf16_wmma<true><<<g, blk, 0, stream>>>(HB, WB, Hb, NN, D3_, DD);
  }
  f32_to_bf16_kernel<<<cdiv((long)DD * NC_, 256), blk, 0, stream>>>(lw3, WB, (long)DD * NC_);
  {
    dim3 g(cdiv(NC_, 64), cdiv(NN, 128));
    gemm_bf16_wmma<false><<<g, blk, 0, stream>>>(HB, WB, RES, NN, NC_, DD);   // res3 [N,40]
  }
  att_scores_kernel<<<cdiv((long)NN * H3_, 8), blk, 0, stream>>>(Hb, a3s, a3d, AS, AD, NN, H3_, NC_);
  fill_f32_kernel<<<cdiv((long)NN * H3_, 256), blk, 0, stream>>>(MX, (long)NN * H3_, -INFINITY);
  fill_f32_kernel<<<cdiv((long)NN * H3_, 256), blk, 0, stream>>>(DEN, (long)NN * H3_, 0.f);
  edge_max_kernel<<<cdiv(EN, 256), blk, 0, stream>>>(ei, AS, AD, MX, EE, NN, H3_);
  edge_exp_kernel<<<cdiv(EN, 256), blk, 0, stream>>>(ei, AS, AD, MX, DEN, P, EE, NN, H3_);
  normalize_alpha_kernel<<<cdiv(EN, 256), blk, 0, stream>>>(ei, P, DEN, nullptr, EE, NN, H3_);
  fill_f32_kernel<<<cdiv((long)NN * D3_, 256), blk, 0, stream>>>(ACC3, (long)NN * D3_, 0.f);
  scatter_msg_kernel<<<EN, blk, 0, stream>>>(ei, Hb, P, ACC3, EE, NN, H3_, NC_);
  add_bias2_kernel<<<cdiv((long)NN * NC_, 256), blk, 0, stream>>>(RES, b3, lb3, (long)NN * NC_, NC_);
  finalize3_kernel<<<cdiv((long)NN * NC_, 256), blk, 0, stream>>>(ACC3, RES, out, NN);
}
